// TGAT_63273458204968
// MI455X (gfx1250) — matrix operand order repaired and checked by reading, not compiled, not run on hardware
//
#include <hip/hip_runtime.h>
#include <stdint.h>

// ---------------------------------------------------------------------------
// TGAT fused kernel for MI455X (gfx1250, wave32, WMMA bf16 16x16x32)
//
// Algebraic refactor: scores = X @ (Wk_h^T q_h), attn_out = (s^T X) @ Wv_h.
// This removes the two (B*N,444)x(444,272) projections (316 GFLOP) and makes
// the kernel HBM-bound (~1.2GB -> ~52us @ 23.3TB/s). Remaining GEMMs are all
// M=16 batched WMMA GEMMs with weights packed to bf16 in a fragment-native
// tiled layout, streamed via global_load_async_to_lds_b128 (double-buffered).
//
// Round-1 fix: wave index is scalarized with readfirstlane so the per-wave
// tile-ownership guards compile to scalar branches (s_cmp/s_cbranch) instead
// of EXEC-masked regions -- WMMA requires EXEC == all-1s (ISA 7.12).
// ---------------------------------------------------------------------------

typedef __attribute__((ext_vector_type(16))) __bf16       v16bf;
typedef __attribute__((ext_vector_type(8)))  float        v8f;
typedef __attribute__((ext_vector_type(8)))  unsigned int v8u;

namespace {

constexpr int B_TOTAL  = 32768;
constexpr int BT       = 16;     // batch rows per block == one WMMA M tile
constexpr int NTHREADS = 256;    // 8 wave32
constexpr int NWAVE    = 8;

// ---- workspace (bf16, chunk-tiled weights) element offsets ----
// layout: dst[(kc*Np + n)*32 + kk], zero-padded, B-fragment native
constexpr int OFF_WQ   = 0;        // Kp=288 Np=272
constexpr int OFF_WKT0 = 78336;    // Kp=160 Np=448 (Wk^T, head 0)
constexpr int OFF_WKT1 = 150016;   // Kp=160 Np=448 (Wk^T, head 1)
constexpr int OFF_WV0  = 221696;   // Kp=448 Np=144
constexpr int OFF_WV1  = 286208;
constexpr int OFF_WR   = 350720;   // Kp=288 Np=272
constexpr int OFF_W1   = 429056;   // Kp=448 Np=176
constexpr int OFF_W2   = 507904;   // Kp=192 Np=176  -> total ~1.06 MB

// ---- LDS byte offsets (single static block, phase-aliased) ----
constexpr int L_QLDS  = 0;        // float[16][288]  query (residual)
constexpr int L_QPROJ = 18432;    // float[16][288]  q; later attn_out
constexpr int L_WQK   = 36864;    // float[16*2][448] wqk; later OLDS f[16][272]
constexpr int L_XB    = 94208;    // float[2][20][448] X dbl-buf; later XMLP f[16][448]
constexpr int L_XBAR  = 165888;   // float[16*2][448] xbar; later H1 f[16][192]
constexpr int L_WBUF  = 223232;   // bf16[2][32*448] weight chunk dbl-buf
constexpr int L_SC    = 280576;   // float[2*20] scores
constexpr int LDS_BYTES = 280832; // ~274 KiB of 320 KiB WGP LDS

constexpr int CHUNK_MAX = 32 * 448;   // elements per weight-chunk buffer

} // namespace

// ---------------------------- CDNA5 async copy -----------------------------
__device__ __forceinline__ unsigned lds_off(const void* p) {
  // generic pointer to LDS: addr[31:0] is the LDS byte offset (ISA 10.2)
  return (unsigned)(uintptr_t)p;
}

__device__ __forceinline__ void async_b128(unsigned ldsoff, const void* g) {
  asm volatile("global_load_async_to_lds_b128 %0, %1, off"
               :: "v"(ldsoff), "v"(g) : "memory");
}

__device__ __forceinline__ void async_wait0() {
#if __has_builtin(__builtin_amdgcn_s_wait_asynccnt)
  __builtin_amdgcn_s_wait_asynccnt(0);
#else
  asm volatile("s_wait_asynccnt 0x0" ::: "memory");
#endif
}

__device__ __forceinline__ void stage_w(__bf16* dst, const __bf16* src,
                                        int nbytes, int tid) {
  unsigned d = lds_off(dst);
  const char* s = (const char*)src;
  for (int o = tid * 16; o < nbytes; o += NTHREADS * 16)
    async_b128(d + (unsigned)o, s + o);
}

// stage 20 rows of X[b] = [nbr_node(172) | nbr_edge(172) | nbr_time(100)] f32
__device__ __forceinline__ void stage_x(const float* nbf, const float* nbe,
                                        const float* nbt, int bg,
                                        unsigned dbase, int tid) {
  const char* f0 = (const char*)(nbf + (size_t)bg * 20 * 172);
  const char* f1 = (const char*)(nbe + (size_t)bg * 20 * 172);
  const char* f2 = (const char*)(nbt + (size_t)bg * 20 * 100);
  for (int j = tid; j < 20 * 111; j += NTHREADS) {   // 43+43+25 b128 per row
    int row = j / 111, q = j - row * 111;
    if (q < 43)
      async_b128(dbase + (unsigned)(row * 1792 + q * 16), f0 + row * 688 + q * 16);
    else if (q < 86) {
      int u = q - 43;
      async_b128(dbase + (unsigned)(row * 1792 + 688 + u * 16), f1 + row * 688 + u * 16);
    } else {
      int u = q - 86;
      async_b128(dbase + (unsigned)(row * 1792 + 1376 + u * 16), f2 + row * 400 + u * 16);
    }
  }
}

// ----------------------------- WMMA fragments ------------------------------
// A (16x32 bf16) from row-major f32 LDS; lane l<16 -> row l, K 0..15;
// lane l>=16 -> row l-16, K 16..31. Zero beyond kmax.
__device__ __forceinline__ v16bf load_a(const float* A, int stride, int k0,
                                        int kmax, int lane) {
  int row = lane & 15;
  int kb  = (lane >> 4) * 16;
  const float* p = A + row * stride;
  v16bf a;
#pragma unroll
  for (int e = 0; e < 8; ++e) {
    int k = k0 + kb + e;
    a[e] = (__bf16)((k < kmax) ? p[k] : 0.0f);
  }
#pragma unroll
  for (int e = 0; e < 8; ++e) {
    int k = k0 + kb + 8 + e;
    a[8 + e] = (__bf16)((k < kmax) ? p[k] : 0.0f);
  }
  return a;
}

// B (32x16 bf16) from chunk-tiled LDS: chunk[(n)*32 + kk].
// lane l<16 -> col l, K 0..15 ; lane l>=16 -> col l-16, K 16..31.
__device__ __forceinline__ v16bf load_b(const __bf16* chunk, int colbase, int lane) {
  int col = colbase + (lane & 15);
  int kh  = (lane >> 4) * 16;
  v8u r = *(const v8u*)(chunk + col * 32 + kh);   // 2x ds_load_b128
  return __builtin_bit_cast(v16bf, r);
}

// -------- generic M=16 GEMM: K chunked by 32, async dbl-buffered W ---------
// `wave` must be an SGPR value (readfirstlane) so tile guards are scalar
// branches and EXEC is all-1s at every WMMA.
template <class ALoad, class Epi>
__device__ __forceinline__ void gemm16(const __bf16* wsW, int Np, int n_chunks,
                                       int n_tiles, __bf16* wbuf, int tid,
                                       int wave, int lane, ALoad aload, Epi epi) {
  v8f acc[4];
#pragma unroll
  for (int i = 0; i < 4; ++i)
#pragma unroll
    for (int e = 0; e < 8; ++e) acc[i][e] = 0.0f;

  const int cbytes = Np * 64;
  stage_w(wbuf, wsW, cbytes, tid);
  async_wait0();
  __syncthreads();

  for (int kc = 0; kc < n_chunks; ++kc) {
    if (kc + 1 < n_chunks)   // prefetch next chunk while computing this one
      stage_w(wbuf + ((kc + 1) & 1) * CHUNK_MAX,
              wsW + (size_t)(kc + 1) * Np * 32, cbytes, tid);

    v16bf a = aload(kc);
    const __bf16* cb = wbuf + (kc & 1) * CHUNK_MAX;
#pragma unroll
    for (int i = 0; i < 4; ++i) {
      int t = wave + i * NWAVE;          // scalar (wave is SGPR)
      if (t < n_tiles) {                 // scalar branch: EXEC stays full
        v16bf bfrag = load_b(cb, t * 16, lane);
        acc[i] = __builtin_amdgcn_wmma_f32_16x16x32_bf16(
            false, a, false, bfrag, (short)0, acc[i], false, false);
      }
    }
    async_wait0();
    __syncthreads();
  }
#pragma unroll
  for (int i = 0; i < 4; ++i) {
    int t = wave + i * NWAVE;
    if (t < n_tiles) epi(t, acc[i]);
  }
  __syncthreads();
}

// ------------------------------ weight packer ------------------------------
__global__ void tgat_pack_w(const float* __restrict__ W, int ldW, int Kdim,
                            int Ndim, int roff, int coff,
                            __bf16* __restrict__ dst, int Kp, int Np,
                            int transposed) {
  int total = Kp * Np;
  for (int idx = blockIdx.x * blockDim.x + threadIdx.x; idx < total;
       idx += gridDim.x * blockDim.x) {
    int kc  = idx / (Np * 32);
    int rem = idx - kc * (Np * 32);
    int n   = rem >> 5;
    int kk  = rem & 31;
    int k   = kc * 32 + kk;
    float v = 0.0f;
    if (k < Kdim && n < Ndim)
      v = transposed ? W[(size_t)n * ldW + roff + k]
                     : W[(size_t)k * ldW + coff + n];
    dst[idx] = (__bf16)v;
  }
}

// ------------------------------- main kernel -------------------------------
__global__ __launch_bounds__(NTHREADS)
void tgat_fused(const float* __restrict__ node_f, const float* __restrict__ node_t,
                const float* __restrict__ nb_f,   const float* __restrict__ nb_t,
                const float* __restrict__ nb_e,   const int* __restrict__ masks,
                const float* __restrict__ br,     const float* __restrict__ ln_g,
                const float* __restrict__ ln_b,   const float* __restrict__ b1,
                const float* __restrict__ b2,     const __bf16* __restrict__ ws,
                float* __restrict__ out) {
  __shared__ __align__(128) char smem[LDS_BYTES];
  float*  QLDS  = (float*)(smem + L_QLDS);    // [16][288] query (residual)
  float*  QPROJ = (float*)(smem + L_QPROJ);   // [16][288] q / attn_out
  float*  WQK   = (float*)(smem + L_WQK);     // [(b*2+h)*448+j] / OLDS[16][272]
  float*  XB    = (float*)(smem + L_XB);      // [2][20][448] / XMLP[16][448]
  float*  XBAR  = (float*)(smem + L_XBAR);    // [(b*2+h)*448+j] / H1[16][192]
  __bf16* WBUF  = (__bf16*)(smem + L_WBUF);
  float*  SC    = (float*)(smem + L_SC);

  const int tid  = threadIdx.x;
  // Scalarized wave id: guarantees tile-ownership guards are scalar branches,
  // keeping EXEC == all-1s at every v_wmma (ISA 7.12 requirement).
  const int wave = __builtin_amdgcn_readfirstlane(tid >> 5);
  const int lane = tid & 31;
  const int bg0  = blockIdx.x * BT;

  // ---- init: zero pads, async-stage query = [node_f | node_time | 0] ----
  for (int i = tid; i < 16 * 16; i += NTHREADS)
    QLDS[(i >> 4) * 288 + 272 + (i & 15)] = 0.0f;
  for (int i = tid; i < 40 * 4; i += NTHREADS)
    XB[(i >> 2) * 448 + 444 + (i & 3)] = 0.0f;
  {
    unsigned d = lds_off(QLDS);
    for (int j = tid; j < 16 * 68; j += NTHREADS) {   // 43 + 25 b128 per row
      int r = j / 68, q = j - r * 68;
      if (q < 43)
        async_b128(d + (unsigned)(r * 1152 + q * 16),
                   (const char*)(node_f + (size_t)(bg0 + r) * 172) + q * 16);
      else {
        int u = q - 43;
        async_b128(d + (unsigned)(r * 1152 + 688 + u * 16),
                   (const char*)(node_t + (size_t)(bg0 + r) * 100) + u * 16);
      }
    }
  }
  async_wait0();
  __syncthreads();

  // ---- GEMM1: q = query @ Wq   (M=16, K=272->288, N=272 = 17 tiles) ----
  gemm16(ws + OFF_WQ, 272, 9, 17, WBUF, tid, wave, lane,
         [&](int kc) { return load_a(QLDS, 288, kc * 32, 272, lane); },
         [&](int t, const v8f& a) {
           int c = t * 16 + (lane & 15), rb = (lane >> 4) << 3;
#pragma unroll
           for (int e = 0; e < 8; ++e) QPROJ[(rb + e) * 288 + c] = a[e];
         });

  // ---- GEMM2: wqk_h = q_h @ Wk_h^T  (K=136->160, N=444->448 = 28 tiles) ----
  for (int h = 0; h < 2; ++h) {
    gemm16(ws + (h ? OFF_WKT1 : OFF_WKT0), 448, 5, 28, WBUF, tid, wave, lane,
           [&](int kc) { return load_a(QPROJ + h * 136, 288, kc * 32, 136, lane); },
           [&](int t, const v8f& a) {
             int c = t * 16 + (lane & 15), rb = (lane >> 4) << 3;
#pragma unroll
             for (int e = 0; e < 8; ++e)
               WQK[((rb + e) * 2 + h) * 448 + c] = a[e];
           });
  }

  // ---- per-element: stream X once; scores, softmax, xbar = s^T X ----
  stage_x(nb_f, nb_e, nb_t, bg0, lds_off(XB), tid);
  async_wait0();
  __syncthreads();
  for (int bl = 0; bl < BT; ++bl) {
    if (bl + 1 < BT)
      stage_x(nb_f, nb_e, nb_t, bg0 + bl + 1,
              lds_off(XB) + (unsigned)(((bl + 1) & 1) * 20 * 448 * 4), tid);
    const float* xb = XB + (bl & 1) * 20 * 448;

    // raw scores: 40 (h,n) pairs, 8 lanes each over j
    for (int p = tid >> 3; p < 40; p += 32) {
      int h = p / 20, n = p - h * 20;
      const float* xr = xb + n * 448;
      const float* wq = WQK + ((size_t)bl * 2 + h) * 448;
      float s = 0.0f;
      for (int j = (tid & 7); j < 448; j += 8) s += xr[j] * wq[j];
      s += __shfl_xor(s, 1); s += __shfl_xor(s, 2); s += __shfl_xor(s, 4);
      if ((tid & 7) == 0) {
        int mk = masks[(size_t)(bg0 + bl) * 20 + n];
        SC[h * 20 + n] = mk ? s * 0.08574929257125441f : -1.0e10f; // 136^-0.5
      }
    }
    __syncthreads();
    if (tid < 2) {                       // softmax over 20, per head
      float m = -3.4e38f;
      for (int n = 0; n < 20; ++n) m = fmaxf(m, SC[tid * 20 + n]);
      float sum = 0.0f, ex[20];
      for (int n = 0; n < 20; ++n) { ex[n] = __expf(SC[tid * 20 + n] - m); sum += ex[n]; }
      float inv = 1.0f / sum;
      for (int n = 0; n < 20; ++n) SC[tid * 20 + n] = ex[n] * inv;
    }
    __syncthreads();
    for (int idx = tid; idx < 2 * 448; idx += NTHREADS) {
      int h = idx / 448, j = idx - h * 448;
      float acc = 0.0f;
      for (int n = 0; n < 20; ++n) acc += SC[h * 20 + n] * xb[n * 448 + j];
      XBAR[((bl * 2 + h) * 448) + j] = acc;
    }
    async_wait0();
    __syncthreads();
  }

  // ---- GEMM3: attn_out_h = xbar_h @ Wv_h  (K=448, N=136->144 = 9 tiles) ----
  for (int h = 0; h < 2; ++h) {
    gemm16(ws + (h ? OFF_WV1 : OFF_WV0), 144, 14, 9, WBUF, tid, wave, lane,
           [&](int kc) { return load_a(XBAR + h * 448, 896, kc * 32, 448, lane); },
           [&](int t, const v8f& a) {
             int c = t * 16 + (lane & 15), rb = (lane >> 4) << 3;
             if (c < 136)
#pragma unroll
               for (int e = 0; e < 8; ++e)
                 QPROJ[(rb + e) * 288 + h * 136 + c] = a[e];
           });
  }

  // ---- GEMM4: out = attn_out @ Wr + br + residual ----
  float* OLDS = WQK;   // [16][272]
  gemm16(ws + OFF_WR, 272, 9, 17, WBUF, tid, wave, lane,
         [&](int kc) { return load_a(QPROJ, 288, kc * 32, 272, lane); },
         [&](int t, const v8f& a) {
           int c = t * 16 + (lane & 15), rb = (lane >> 4) << 3;
#pragma unroll
           for (int e = 0; e < 8; ++e) {
             int r = rb + e;
             OLDS[r * 272 + c] = a[e] + br[c] + QLDS[r * 288 + c];
           }
         });

  // ---- LayerNorm + build x = [LN(out) | node_f | 0] ----
  float* XMLP = XB;    // [16][448]
  {
    int r = wave * 2 + (lane >> 4);      // 8 waves x 2 half-wave rows
    int l = lane & 15;
    float s = 0.0f, s2 = 0.0f;
    for (int c = l; c < 272; c += 16) {
      float v = OLDS[r * 272 + c]; s += v; s2 += v * v;
    }
#pragma unroll
    for (int off = 1; off < 16; off <<= 1) {
      s += __shfl_xor(s, off); s2 += __shfl_xor(s2, off);
    }
    float mu  = s * (1.0f / 272.0f);
    float var = s2 * (1.0f / 272.0f) - mu * mu;
    float inv = rsqrtf(var + 1e-5f);
    for (int c = l; c < 272; c += 16) {
      float v = (OLDS[r * 272 + c] - mu) * inv;
      XMLP[r * 448 + c] = v * ln_g[c] + ln_b[c];
    }
    for (int c = 272 + l; c < 448; c += 16)
      XMLP[r * 448 + c] = (c < 444) ? node_f[(size_t)(bg0 + r) * 172 + (c - 272)] : 0.0f;
  }
  __syncthreads();

  // ---- GEMM5: h1 = relu(x @ W1 + b1)  (K=444->448, N=172->176 = 11 tiles) ----
  float* H1 = XBAR;    // [16][192]
  gemm16(ws + OFF_W1, 176, 14, 11, WBUF, tid, wave, lane,
         [&](int kc) { return load_a(XMLP, 448, kc * 32, 444, lane); },
         [&](int t, const v8f& a) {
           int c = t * 16 + (lane & 15), rb = (lane >> 4) << 3;
#pragma unroll
           for (int e = 0; e < 8; ++e) {
             int r = rb + e;
             H1[r * 192 + c] = (c < 172) ? fmaxf(a[e] + b1[c], 0.0f) : 0.0f;
           }
         });

  // ---- GEMM6: out = h1 @ W2 + b2 -> global ----
  gemm16(ws + OFF_W2, 176, 6, 11, WBUF, tid, wave, lane,
         [&](int kc) { return load_a(H1, 192, kc * 32, 176, lane); },
         [&](int t, const v8f& a) {
           int c = t * 16 + (lane & 15), rb = (lane >> 4) << 3;
           if (c < 172)
#pragma unroll
             for (int e = 0; e < 8; ++e) {
               int r = rb + e;
               out[(size_t)(bg0 + r) * 172 + c] = a[e] + b2[c];
             }
         });
}

// -------------------------------- launcher ---------------------------------
extern "C" void kernel_launch(void* const* d_in, const int* in_sizes, int n_in,
                              void* d_out, int out_size, void* d_ws, size_t ws_size,
                              hipStream_t stream) {
  const float* node_f = (const float*)d_in[0];
  const float* node_t = (const float*)d_in[1];
  const float* nb_f   = (const float*)d_in[2];
  const float* nb_t   = (const float*)d_in[3];
  const float* nb_e   = (const float*)d_in[4];
  const int*   masks  = (const int*)  d_in[5];
  const float* Wq = (const float*)d_in[6];
  const float* Wk = (const float*)d_in[7];
  const float* Wv = (const float*)d_in[8];
  const float* Wr = (const float*)d_in[9];
  const float* br = (const float*)d_in[10];
  const float* lg = (const float*)d_in[11];
  const float* lb = (const float*)d_in[12];
  const float* W1 = (const float*)d_in[13];
  const float* b1 = (const float*)d_in[14];
  const float* W2 = (const float*)d_in[15];
  const float* b2 = (const float*)d_in[16];
  __bf16* ws = (__bf16*)d_ws;   // ~1.06 MB of bf16 packed weights

  dim3 pb(256), pg(256);
  // (W, ldW, Kdim, Ndim, roff, coff, dst, Kp, Np, transposed)
  tgat_pack_w<<<pg, pb, 0, stream>>>(Wq, 272, 272, 272, 0, 0,   ws + OFF_WQ,   288, 272, 0);
  tgat_pack_w<<<pg, pb, 0, stream>>>(Wk, 272, 136, 444, 0, 0,   ws + OFF_WKT0, 160, 448, 1);
  tgat_pack_w<<<pg, pb, 0, stream>>>(Wk, 272, 136, 444, 136, 0, ws + OFF_WKT1, 160, 448, 1);
  tgat_pack_w<<<pg, pb, 0, stream>>>(Wv, 272, 444, 136, 0, 0,   ws + OFF_WV0,  448, 144, 0);
  tgat_pack_w<<<pg, pb, 0, stream>>>(Wv, 272, 444, 136, 0, 136, ws + OFF_WV1,  448, 144, 0);
  tgat_pack_w<<<pg, pb, 0, stream>>>(Wr, 272, 272, 272, 0, 0,   ws + OFF_WR,   288, 272, 0);
  tgat_pack_w<<<pg, pb, 0, stream>>>(W1, 172, 444, 172, 0, 0,   ws + OFF_W1,   448, 176, 0);
  tgat_pack_w<<<pg, pb, 0, stream>>>(W2, 172, 172, 172, 0, 0,   ws + OFF_W2,   192, 176, 0);

  tgat_fused<<<dim3(B_TOTAL / BT), dim3(NTHREADS), 0, stream>>>(
      node_f, node_t, nb_f, nb_t, nb_e, masks, br, lg, lb, b1, b2, ws,
      (float*)d_out);
}
